// CrossShapedSpatialAttention_18013092839823
// MI455X (gfx1250) — matrix operand order repaired and checked by reading, not compiled, not run on hardware
//
#include <hip/hip_runtime.h>
#include <hip/hip_bf16.h>

typedef __bf16 bf16;
typedef __attribute__((ext_vector_type(16))) __bf16 v16bf;
typedef __attribute__((ext_vector_type(8)))  __bf16 v8bf;
typedef __attribute__((ext_vector_type(4)))  __bf16 v4bf;
typedef __attribute__((ext_vector_type(8)))  float   v8f;
typedef __attribute__((ext_vector_type(2)))  int     i32x2;
typedef __attribute__((ext_vector_type(4)))  int     i32x4;

#define NPIX   25088      // 8*56*56
#define QKVW   384
#define LSTR   392        // stripe length 7*56
#define SCALE  0.17677669529663687f   // 32^-0.5

#define GLOBAL_AS __attribute__((address_space(1)))
#define LDS_AS    __attribute__((address_space(3)))

// ---- gfx1250 async global->LDS copy (ASYNCcnt-tracked), with safe fallback ----
#if __has_builtin(__builtin_amdgcn_global_load_async_to_lds_b64)
#define HAVE_ASYNC_LDS 1
static __device__ inline void async_ld_b64(const void* g, void* l) {
  __builtin_amdgcn_global_load_async_to_lds_b64((GLOBAL_AS i32x2*)g, (LDS_AS i32x2*)l, 0, 0);
}
#else
#define HAVE_ASYNC_LDS 0
#endif

#if __has_builtin(__builtin_amdgcn_global_load_async_to_lds_b128)
#define HAVE_ASYNC_LDS128 1
static __device__ inline void async_ld_b128(const void* g, void* l) {
  __builtin_amdgcn_global_load_async_to_lds_b128((GLOBAL_AS i32x4*)g, (LDS_AS i32x4*)l, 0, 0);
}
#else
#define HAVE_ASYNC_LDS128 0
#endif

#if __has_builtin(__builtin_amdgcn_s_wait_asynccnt)
#define WAIT_ASYNC() __builtin_amdgcn_s_wait_asynccnt(0)
#else
#define WAIT_ASYNC() asm volatile("s_wait_asynccnt 0" ::: "memory")
#endif

static __device__ inline v16bf combine(v8bf lo, v8bf hi) {
  union { v16bf v; v8bf h[2]; } u;
  u.h[0] = lo; u.h[1] = hi;
  return u.v;
}

static __device__ inline v8f wmma_bf16(v16bf a, v16bf b, v8f c) {
  return __builtin_amdgcn_wmma_f32_16x16x32_bf16(false, a, false, b, (short)0, c, false, false);
}

// pixel index of position l within stripe s for direction dir (0=horizontal,1=vertical)
static __device__ inline int pix_of(int dir, int s, int l) {
  if (dir == 0) return s * LSTR + l;          // horizontal stripes are contiguous
  int b = s >> 3, mv = s & 7;
  int ww = l / 56, h = l % 56;                // l = ww*H + h
  return b * 3136 + h * 56 + mv * 7 + ww;
}

// ---------------------------------------------------------------------------
// Kernel 1: QKV projection, both directions.  C[M=25088, N=384] = A[M,128]@W[128,384]+b
// block: 128 threads (4 waves); each block computes a 64x64 tile; wave -> 16x64.
// (f32 -> bf16 conversion means these fills must stay on the VALU path.)
// ---------------------------------------------------------------------------
__global__ __launch_bounds__(128)
void qkv_kernel(const float* __restrict__ x,
                const float* __restrict__ Wh, const float* __restrict__ bh,
                const float* __restrict__ Wv, const float* __restrict__ bv,
                bf16* __restrict__ qkv_h, bf16* __restrict__ qkv_v) {
  const int mt = blockIdx.x, ntb = blockIdx.y, dir = blockIdx.z;
  const float* Wq = dir ? Wv : Wh;
  const float* bq = dir ? bv : bh;
  bf16* out = dir ? qkv_v : qkv_h;

  __shared__ __align__(16) bf16 As[64][32];
  __shared__ __align__(16) bf16 Bs[64][32];   // transposed: [n][k]

  const int t = threadIdx.x;
  const int lane = t & 31, wv = t >> 5;
  const int hl = lane >> 4, l16 = lane & 15;

  v8f acc[4] = {};

  for (int kk = 0; kk < 128; kk += 32) {
    __syncthreads();
    // A tile: 64 rows x 32 k, f32 -> bf16
#pragma unroll
    for (int i = 0; i < 4; ++i) {
      int lin = (i * 128 + t) * 4;
      int r = lin >> 5, c = lin & 31;
      const float4 f = *(const float4*)(x + (size_t)(mt * 64 + r) * 256 + dir * 128 + kk + c);
      v4bf h; h.x = (bf16)f.x; h.y = (bf16)f.y; h.z = (bf16)f.z; h.w = (bf16)f.w;
      *(v4bf*)&As[r][c] = h;
    }
    // B tile: 32 k x 64 n, stored transposed
#pragma unroll
    for (int i = 0; i < 4; ++i) {
      int lin = (i * 128 + t) * 4;
      int r = lin >> 6, c = lin & 63;
      const float4 f = *(const float4*)(Wq + (size_t)(kk + r) * QKVW + ntb * 64 + c);
      Bs[c + 0][r] = (bf16)f.x; Bs[c + 1][r] = (bf16)f.y;
      Bs[c + 2][r] = (bf16)f.z; Bs[c + 3][r] = (bf16)f.w;
    }
    __syncthreads();

    const int arow = wv * 16 + l16;
    v16bf a = combine(*(const v8bf*)&As[arow][hl * 8],
                      *(const v8bf*)&As[arow][16 + hl * 8]);
#pragma unroll
    for (int n = 0; n < 4; ++n) {
      int bc = n * 16 + l16;
      v16bf b = combine(*(const v8bf*)&Bs[bc][hl * 8],
                        *(const v8bf*)&Bs[bc][16 + hl * 8]);
      acc[n] = wmma_bf16(a, b, acc[n]);
    }
  }

#pragma unroll
  for (int n = 0; n < 4; ++n)
#pragma unroll
    for (int r = 0; r < 8; ++r) {
      int row = mt * 64 + wv * 16 + hl * 8 + r;
      int col = ntb * 64 + n * 16 + l16;
      out[(size_t)row * QKVW + col] = (bf16)(acc[n][r] + bq[col]);
    }
}

// ---------------------------------------------------------------------------
// Kernel 2: flash attention per (stripe, dir, head).
// block: 256 threads = 8 waves; wave owns query tiles {w, w+8, w+16, w+24}.
// keys processed in chunks of 32 (392 padded to 416).
// K tile fills use async global->LDS DMA when available (straight bf16 copy);
// V needs a transpose so it stays on the VALU path.
// ---------------------------------------------------------------------------
__global__ __launch_bounds__(256)
void attn_kernel(const bf16* __restrict__ qkv_h, const bf16* __restrict__ qkv_v,
                 bf16* __restrict__ o_ws) {
  const int s  = blockIdx.x;
  const int hz = blockIdx.y;
  const int dir = hz >> 2, head = hz & 3;
  const bf16* qkv = dir ? qkv_v : qkv_h;

  const int t = threadIdx.x, lane = t & 31, wv = t >> 5;
  const int hl = lane >> 4, l16 = lane & 15;

  __shared__ __align__(16) bf16 Ks[32][32];     // [key][dim]
  __shared__ __align__(16) bf16 Vt[32][32];     // [dim][key]  (transposed V)
  __shared__ __align__(16) bf16 Ps[8][16][32];  // per-wave P staging

  const int qoff = head * 32, koff = 128 + head * 32, voff = 256 + head * 32;

  // Q fragments (A layout), loaded once
  v16bf qfrag[4];
#pragma unroll
  for (int i = 0; i < 4; ++i) {
    int qt = wv + i * 8;
    int lq = qt * 16 + l16; if (lq > LSTR - 1) lq = LSTR - 1;
    size_t base = (size_t)pix_of(dir, s, lq) * QKVW + qoff;
    qfrag[i] = combine(*(const v8bf*)(qkv + base + hl * 8),
                       *(const v8bf*)(qkv + base + 16 + hl * 8));
  }

  v8f acc[4][2] = {};
  float mrow[4][8], lrow[4][8];
#pragma unroll
  for (int i = 0; i < 4; ++i)
#pragma unroll
    for (int r = 0; r < 8; ++r) { mrow[i][r] = -3.0e38f; lrow[i][r] = 0.0f; }

  for (int ks = 0; ks < 416; ks += 32) {
    __syncthreads();
    {
      int key = t >> 3, c4 = (t & 7) * 4;
      int lk = ks + key; if (lk > LSTR - 1) lk = LSTR - 1;
      size_t base = (size_t)pix_of(dir, s, lk) * QKVW;
#if HAVE_ASYNC_LDS
      async_ld_b64(qkv + base + koff + c4, &Ks[key][c4]);   // ASYNCcnt DMA fill
#else
      *(v4bf*)&Ks[key][c4] = *(const v4bf*)(qkv + base + koff + c4);
#endif
      v4bf vd = *(const v4bf*)(qkv + base + voff + c4);
      Vt[c4 + 0][key] = vd.x; Vt[c4 + 1][key] = vd.y;
      Vt[c4 + 2][key] = vd.z; Vt[c4 + 3][key] = vd.w;
#if HAVE_ASYNC_LDS
      WAIT_ASYNC();
#endif
    }
    __syncthreads();

    // K / V fragments are shared across this wave's 4 query tiles: hoist.
    v16bf kfrag[2], vfrag[2];
#pragma unroll
    for (int n = 0; n < 2; ++n) {
      int kc = n * 16 + l16;
      kfrag[n] = combine(*(const v8bf*)&Ks[kc][hl * 8],
                         *(const v8bf*)&Ks[kc][16 + hl * 8]);
      int dc = n * 16 + l16;
      vfrag[n] = combine(*(const v8bf*)&Vt[dc][hl * 8],
                         *(const v8bf*)&Vt[dc][16 + hl * 8]);
    }

#pragma unroll
    for (int i = 0; i < 4; ++i) {
      // scores: Q (16x32) x K^T (32x32) as two 16x16 tiles
      v8f sc[2];
#pragma unroll
      for (int n = 0; n < 2; ++n) {
        v8f z = {};
        sc[n] = wmma_bf16(qfrag[i], kfrag[n], z);
      }
#pragma unroll
      for (int n = 0; n < 2; ++n) {
        bool valid = (ks + n * 16 + l16) < LSTR;
#pragma unroll
        for (int r = 0; r < 8; ++r)
          sc[n][r] = valid ? sc[n][r] * SCALE : -3.0e38f;
      }
      // online softmax (row = hl*8 + r, spread across 16 lanes)
#pragma unroll
      for (int r = 0; r < 8; ++r) {
        float mx = fmaxf(sc[0][r], sc[1][r]);
        for (int d = 1; d < 16; d <<= 1) mx = fmaxf(mx, __shfl_xor(mx, d));
        float mnew = fmaxf(mrow[i][r], mx);
        float alpha = __expf(mrow[i][r] - mnew);
        float p0 = __expf(sc[0][r] - mnew);
        float p1 = __expf(sc[1][r] - mnew);
        float psum = p0 + p1;
        for (int d = 1; d < 16; d <<= 1) psum += __shfl_xor(psum, d);
        lrow[i][r] = lrow[i][r] * alpha + psum;
        mrow[i][r] = mnew;
        acc[i][0][r] *= alpha; acc[i][1][r] *= alpha;
        sc[0][r] = p0; sc[1][r] = p1;
      }
      // stage P (C layout -> A layout) through per-wave LDS
#pragma unroll
      for (int n = 0; n < 2; ++n)
#pragma unroll
        for (int r = 0; r < 8; ++r)
          Ps[wv][hl * 8 + r][n * 16 + l16] = (bf16)sc[n][r];
      v16bf pf = combine(*(const v8bf*)&Ps[wv][l16][hl * 8],
                         *(const v8bf*)&Ps[wv][l16][16 + hl * 8]);
      // accumulate P (16x32) x V (32x32)
#pragma unroll
      for (int n = 0; n < 2; ++n)
        acc[i][n] = wmma_bf16(pf, vfrag[n], acc[i][n]);
    }
  }

  // epilogue: o_ws[pixel][dir*128 + head*32 + d] = acc / l
#pragma unroll
  for (int i = 0; i < 4; ++i) {
    int qt = wv + i * 8;
    if (qt >= 25) continue;
#pragma unroll
    for (int r = 0; r < 8; ++r) {
      int lq = qt * 16 + hl * 8 + r;
      if (lq >= LSTR) continue;
      float inv = 1.0f / lrow[i][r];
      size_t pix = (size_t)pix_of(dir, s, lq);
#pragma unroll
      for (int n = 0; n < 2; ++n)
        o_ws[pix * 256 + dir * 128 + head * 32 + n * 16 + l16] =
            (bf16)(acc[i][n][r] * inv);
    }
  }
}

// ---------------------------------------------------------------------------
// Kernel 3: projection.  out[M=25088, 256] = o_ws[M,256] @ W_proj[256,256] + b
// A tile is a straight bf16 copy -> async global->LDS DMA when available.
// ---------------------------------------------------------------------------
__global__ __launch_bounds__(128)
void proj_kernel(const bf16* __restrict__ o_ws,
                 const float* __restrict__ Wp, const float* __restrict__ bp,
                 float* __restrict__ out) {
  const int mt = blockIdx.x, ntb = blockIdx.y;
  __shared__ __align__(16) bf16 As[64][32];
  __shared__ __align__(16) bf16 Bs[64][32];

  const int t = threadIdx.x, lane = t & 31, wv = t >> 5;
  const int hl = lane >> 4, l16 = lane & 15;

  v8f acc[4] = {};

  for (int kk = 0; kk < 256; kk += 32) {
    __syncthreads();
#pragma unroll
    for (int i = 0; i < 2; ++i) {
      int lin = (i * 128 + t) * 8;
      int r = lin >> 5, c = lin & 31;
#if HAVE_ASYNC_LDS128
      async_ld_b128(o_ws + (size_t)(mt * 64 + r) * 256 + kk + c, &As[r][c]);
#else
      *(v8bf*)&As[r][c] = *(const v8bf*)(o_ws + (size_t)(mt * 64 + r) * 256 + kk + c);
#endif
    }
#pragma unroll
    for (int i = 0; i < 4; ++i) {
      int lin = (i * 128 + t) * 4;
      int r = lin >> 6, c = lin & 63;
      const float4 f = *(const float4*)(Wp + (size_t)(kk + r) * 256 + ntb * 64 + c);
      Bs[c + 0][r] = (bf16)f.x; Bs[c + 1][r] = (bf16)f.y;
      Bs[c + 2][r] = (bf16)f.z; Bs[c + 3][r] = (bf16)f.w;
    }
#if HAVE_ASYNC_LDS128
    WAIT_ASYNC();
#endif
    __syncthreads();

    const int arow = wv * 16 + l16;
    v16bf a = combine(*(const v8bf*)&As[arow][hl * 8],
                      *(const v8bf*)&As[arow][16 + hl * 8]);
#pragma unroll
    for (int n = 0; n < 4; ++n) {
      int bc = n * 16 + l16;
      v16bf b = combine(*(const v8bf*)&Bs[bc][hl * 8],
                        *(const v8bf*)&Bs[bc][16 + hl * 8]);
      acc[n] = wmma_bf16(a, b, acc[n]);
    }
  }

#pragma unroll
  for (int n = 0; n < 4; ++n)
#pragma unroll
    for (int r = 0; r < 8; ++r) {
      int row = mt * 64 + wv * 16 + hl * 8 + r;
      int col = ntb * 64 + n * 16 + l16;
      out[(size_t)row * 256 + col] = acc[n][r] + bp[col];
    }
}

// ---------------------------------------------------------------------------
extern "C" void kernel_launch(void* const* d_in, const int* in_sizes, int n_in,
                              void* d_out, int out_size, void* d_ws, size_t ws_size,
                              hipStream_t stream) {
  const float* x  = (const float*)d_in[0];
  const float* Wh = (const float*)d_in[1];
  const float* bh = (const float*)d_in[2];
  const float* Wv = (const float*)d_in[3];
  const float* bv = (const float*)d_in[4];
  const float* Wp = (const float*)d_in[5];
  const float* bp = (const float*)d_in[6];

  bf16* qh = (bf16*)d_ws;                               // 25088 x 384
  bf16* qv = qh + (size_t)NPIX * QKVW;                  // 25088 x 384
  bf16* ow = qv + (size_t)NPIX * QKVW;                  // 25088 x 256

  qkv_kernel<<<dim3(NPIX / 64, QKVW / 64, 2), 128, 0, stream>>>(x, Wh, bh, Wv, bv, qh, qv);
  attn_kernel<<<dim3(64, 8), 256, 0, stream>>>(qh, qv, ow);
  proj_kernel<<<dim3(NPIX / 64, 4), 128, 0, stream>>>(ow, Wp, bp, (float*)d_out);
}
